// MeToken_27745488732424
// MI455X (gfx1250) — matrix operand order repaired and checked by reading, not compiled, not run on hardware
//
#include <hip/hip_runtime.h>
#include <hip/hip_bf16.h>

// ---------------------------------------------------------------------------
// MeToken grouped VQ + uniform loss for MI455X (gfx1250, wave32, WMMA)
// Strategy: bin tokens by PTM type, run dense per-type GEMMs through
// v_wmma_f32_16x16x32_bf16 (bf16 in, f32 accumulate), fuse softmax/argmax and
// the Gram-matrix contrastive loss row-wise.
// ---------------------------------------------------------------------------

typedef __attribute__((ext_vector_type(16))) __bf16 v16bf;
typedef __attribute__((ext_vector_type(8)))  __bf16 v8bf;
typedef __attribute__((ext_vector_type(8)))  float  v8f;

#define N_TOK   8192
#define DIM     256
#define NTYPES  26
#define KPT     128                 // codes per type
#define NCODES  (NTYPES * KPT)      // 3328
#define ROWTILES (NCODES / 16)      // 208
#define COLCHUNKS 8
#define TILES_PER_CHUNK (ROWTILES / COLCHUNKS) // 26
#define MAXTILES 512                // ceil(N_TOK/16) upper bound per type

static __device__ __forceinline__ v16bf join16(v8bf lo, v8bf hi) {
  v16bf r;
#pragma unroll
  for (int i = 0; i < 8; ++i) { r[i] = lo[i]; r[i + 8] = hi[i]; }
  return r;
}

// ---------------- binning ----------------
__global__ void k_init(int* counts, int* cursor) {
  int i = threadIdx.x;
  if (i < 32) { counts[i] = 0; cursor[i] = 0; }
}

__global__ void k_hist(const int* __restrict__ Q, int* __restrict__ counts) {
  int n = blockIdx.x * blockDim.x + threadIdx.x;
  if (n < N_TOK) atomicAdd(&counts[Q[n]], 1);
}

__global__ void k_scan(const int* __restrict__ counts, int* __restrict__ offsets,
                       int* __restrict__ cursor) {
  if (threadIdx.x == 0) {
    int acc = 0;
    for (int t = 0; t < NTYPES; ++t) { offsets[t] = acc; cursor[t] = acc; acc += counts[t]; }
  }
}

__global__ void k_scatter(const int* __restrict__ Q, int* __restrict__ cursor,
                          int* __restrict__ perm) {
  int n = blockIdx.x * blockDim.x + threadIdx.x;
  if (n < N_TOK) { int p = atomicAdd(&cursor[Q[n]], 1); perm[p] = n; }
}

// ---------------- precision conversion ----------------
__global__ void k_cvt_x(const float* __restrict__ x, __bf16* __restrict__ xbf) {
  long i = (long)blockIdx.x * blockDim.x + threadIdx.x;
  if (i < (long)N_TOK * DIM) xbf[i] = (__bf16)x[i];
}

// one block per codebook row: raw bf16, normalized bf16, raw transposed bf16
__global__ void k_cvt_E(const float* __restrict__ E, __bf16* __restrict__ Ebf,
                        __bf16* __restrict__ Enbf, __bf16* __restrict__ EbfT) {
  __shared__ float red[DIM];
  int r = blockIdx.x, d = threadIdx.x;
  float v = E[(long)r * DIM + d];
  red[d] = v * v;
  __syncthreads();
  for (int s = DIM / 2; s > 0; s >>= 1) {
    if (d < s) red[d] += red[d + s];
    __syncthreads();
  }
  float rn = rsqrtf(red[0] + 1e-30f);
  Ebf[(long)r * DIM + d]  = (__bf16)v;
  Enbf[(long)r * DIM + d] = (__bf16)(v * rn);
  EbfT[(long)d * NCODES + r] = (__bf16)v;
}

// ---------------- grouped quantize: one wave per 16-token tile ----------------
__global__ __launch_bounds__(32) void k_quant(
    const __bf16* __restrict__ xbf, const __bf16* __restrict__ Ebf,
    const __bf16* __restrict__ EbfT, const int* __restrict__ counts,
    const int* __restrict__ offsets, const int* __restrict__ perm,
    const void* __restrict__ tau_p, float* __restrict__ out) {
  __shared__ __bf16 sims[16 * KPT];           // 4 KB: sim tile, C-layout -> A-layout staging

  int t = blockIdx.x / MAXTILES;
  int tile = blockIdx.x % MAXTILES;
  int cnt = counts[t];
  if (tile * 16 >= cnt) return;
  int off = offsets[t];

  int lane = threadIdx.x;
  int half = lane >> 4, l16 = lane & 15;

  // tau: scalar may arrive as int32 or float32 bits — sniff
  int iv = *(const int*)tau_p;
  float fv = __int_as_float(iv);
  float af = fabsf(fv);
  float tau = (af > 1e-12f && af < 1e12f) ? fv : (float)iv;
  float invTau = 1.0f / tau;

  // A operand: 16 token rows in WMMA 16-bit A(16x32) layout, 8 K-chunks resident
  int mA = tile * 16 + l16;
  int tokA = perm[off + (mA < cnt ? mA : cnt - 1)];
  const __bf16* xr = xbf + (long)tokA * DIM;
  v16bf a[8];
#pragma unroll
  for (int kc = 0; kc < 8; ++kc) {
    int base = kc * 32 + half * 16;
    a[kc] = join16(*(const v8bf*)(xr + base), *(const v8bf*)(xr + base + 8));
  }

  // dots[16 tokens x 128 codes] = X * cb^T
  v8f acc[8];
#pragma unroll
  for (int ct = 0; ct < 8; ++ct) {
    v8f c = {};
    const __bf16* er = Ebf + (long)(t * KPT + ct * 16 + l16) * DIM + half * 16;
#pragma unroll
    for (int kc = 0; kc < 8; ++kc) {
      v16bf b = *(const v16bf*)(er + kc * 32);
      c = __builtin_amdgcn_wmma_f32_16x16x32_bf16(false, a[kc], false, b,
                                                  (short)0, c, false, false);
    }
    acc[ct] = c;
  }

  // row-wise softmax + argmax (row M = r + 8*half lives across 16 lanes of a half)
  int tokR[8]; int valR[8];
#pragma unroll
  for (int r = 0; r < 8; ++r) {
    int mm = tile * 16 + r + half * 8;
    valR[r] = (mm < cnt);
    tokR[r] = perm[off + (mm < cnt ? mm : cnt - 1)];
  }
#pragma unroll
  for (int r = 0; r < 8; ++r) {
    float v[8];
#pragma unroll
    for (int ct = 0; ct < 8; ++ct) v[ct] = acc[ct][r] * invTau;
    float mx = v[0]; int mcode = l16;
#pragma unroll
    for (int ct = 1; ct < 8; ++ct)
      if (v[ct] > mx) { mx = v[ct]; mcode = ct * 16 + l16; }
#pragma unroll
    for (int d = 1; d < 16; d <<= 1) {
      float om = __shfl_xor(mx, d);
      int   oc = __shfl_xor(mcode, d);
      if (om > mx || (om == mx && oc < mcode)) { mx = om; mcode = oc; }
    }
    float e[8]; float s = 0.f;
#pragma unroll
    for (int ct = 0; ct < 8; ++ct) { e[ct] = expf(v[ct] - mx); s += e[ct]; }
#pragma unroll
    for (int d = 1; d < 16; d <<= 1) s += __shfl_xor(s, d);
    float inv = 1.0f / s;
    int row = r + half * 8;
#pragma unroll
    for (int ct = 0; ct < 8; ++ct)
      sims[row * KPT + ct * 16 + l16] = (__bf16)(e[ct] * inv);
    if (l16 == 0 && valR[r])
      out[(long)N_TOK * DIM + 2 + tokR[r]] = (float)(t * KPT + mcode);
  }
  __syncthreads();

  // z = sim(16x128) * cb(128x256); sim reloaded from LDS in A layout
  v16bf sa[4];
#pragma unroll
  for (int kc = 0; kc < 4; ++kc) {
    int base = kc * 32 + half * 16;
    sa[kc] = join16(*(const v8bf*)(&sims[l16 * KPT + base]),
                    *(const v8bf*)(&sims[l16 * KPT + base + 8]));
  }
#pragma unroll
  for (int dt = 0; dt < 16; ++dt) {
    v8f z = {};
    const __bf16* bt = EbfT + (long)(dt * 16 + l16) * NCODES + t * KPT + half * 16;
#pragma unroll
    for (int kc = 0; kc < 4; ++kc) {
      v16bf b = *(const v16bf*)(bt + kc * 32);
      z = __builtin_amdgcn_wmma_f32_16x16x32_bf16(false, sa[kc], false, b,
                                                  (short)0, z, false, false);
    }
    int d = dt * 16 + l16;
#pragma unroll
    for (int r = 0; r < 8; ++r)
      if (valR[r]) out[(long)tokR[r] * DIM + d] = z[r];
  }
}

// ---------------- uniform loss: one wave per (16-row tile, column chunk) ----------------
__global__ __launch_bounds__(32) void k_uniform(const __bf16* __restrict__ Enbf,
                                                float* __restrict__ pSum,
                                                float* __restrict__ pPos) {
  int rowTile = blockIdx.x;     // 0..207
  int chunk   = blockIdx.y;     // 0..7
  int lane = threadIdx.x, half = lane >> 4, l16 = lane & 15;
  int i0 = rowTile * 16;

  const __bf16* ar = Enbf + (long)(i0 + l16) * DIM;
  v16bf a[8];
#pragma unroll
  for (int kc = 0; kc < 8; ++kc) {
    int base = kc * 32 + half * 16;
    a[kc] = join16(*(const v8bf*)(ar + base), *(const v8bf*)(ar + base + 8));
  }

  float sAll[8] = {0, 0, 0, 0, 0, 0, 0, 0};
  float sPos[8] = {0, 0, 0, 0, 0, 0, 0, 0};
  const float invT = 1.0f / 0.07f;

  for (int jt = chunk * TILES_PER_CHUNK; jt < (chunk + 1) * TILES_PER_CHUNK; ++jt) {
    v8f c = {};
    const __bf16* br = Enbf + (long)(jt * 16 + l16) * DIM + half * 16;
#pragma unroll
    for (int kc = 0; kc < 8; ++kc) {
      v16bf b = *(const v16bf*)(br + kc * 32);
      c = __builtin_amdgcn_wmma_f32_16x16x32_bf16(false, a[kc], false, b,
                                                  (short)0, c, false, false);
    }
    int col = jt * 16 + l16;
#pragma unroll
    for (int r = 0; r < 8; ++r) {
      int i = i0 + r + half * 8;
      if (col != i) {                       // diagonal -> -inf -> exp 0
        float e = expf(c[r] * invT);
        sAll[r] += e;
        if ((col >> 7) == (i >> 7)) sPos[r] += e;   // same PTM type (128 = 2^7)
      }
    }
  }
#pragma unroll
  for (int r = 0; r < 8; ++r) {
#pragma unroll
    for (int d = 1; d < 16; d <<= 1) {
      sAll[r] += __shfl_xor(sAll[r], d);
      sPos[r] += __shfl_xor(sPos[r], d);
    }
    if (l16 == 0) {
      int i = i0 + r + half * 8;
      pSum[i * COLCHUNKS + chunk] = sAll[r];  // unique writer -> deterministic
      pPos[i * COLCHUNKS + chunk] = sPos[r];
    }
  }
}

__global__ void k_final(const float* __restrict__ pSum, const float* __restrict__ pPos,
                        float* __restrict__ out) {
  __shared__ float red[256];
  int tid = threadIdx.x;
  float acc = 0.f;
  for (int i = tid; i < NCODES; i += 256) {
    float s = 0.f, p = 0.f;
    for (int c = 0; c < COLCHUNKS; ++c) { s += pSum[i * COLCHUNKS + c]; p += pPos[i * COLCHUNKS + c]; }
    acc += logf(p / s);
  }
  red[tid] = acc;
  __syncthreads();
  for (int s = 128; s > 0; s >>= 1) {
    if (tid < s) red[tid] += red[tid + s];
    __syncthreads();
  }
  if (tid == 0) {
    out[(long)N_TOK * DIM]     = 0.0f;                  // loss
    out[(long)N_TOK * DIM + 1] = -red[0] / (float)NCODES; // uniform_loss
  }
}

// ---------------- host-side launch ----------------
extern "C" void kernel_launch(void* const* d_in, const int* in_sizes, int n_in,
                              void* d_out, int out_size, void* d_ws, size_t ws_size,
                              hipStream_t stream) {
  (void)in_sizes; (void)n_in; (void)out_size; (void)ws_size;

  const float* x = (const float*)d_in[0];
  const int*   Q = (const int*)d_in[1];
  const float* E = (const float*)d_in[2];
  const void*  tau = d_in[3];
  float* out = (float*)d_out;

  char* ws = (char*)d_ws;
  int* counts  = (int*)(ws + 0);
  int* offsets = (int*)(ws + 128);
  int* cursor  = (int*)(ws + 256);
  int* perm    = (int*)(ws + 384);                       // 8192 ints

  const size_t SZ_E = (size_t)NCODES * DIM * 2;          // 1,703,936 B
  __bf16* xbf  = (__bf16*)(ws + 65536);                  // 4 MB
  __bf16* Ebf  = (__bf16*)(ws + 65536 + (size_t)N_TOK * DIM * 2);
  __bf16* Enbf = (__bf16*)((char*)Ebf + SZ_E);
  __bf16* EbfT = (__bf16*)((char*)Enbf + SZ_E);
  float*  pSum = (float*)((char*)EbfT + SZ_E);           // NCODES * 8 floats
  float*  pPos = (float*)((char*)pSum + (size_t)NCODES * COLCHUNKS * 4);

  k_init   <<<1, 32, 0, stream>>>(counts, cursor);
  k_hist   <<<N_TOK / 256, 256, 0, stream>>>(Q, counts);
  k_scan   <<<1, 32, 0, stream>>>(counts, offsets, cursor);
  k_scatter<<<N_TOK / 256, 256, 0, stream>>>(Q, cursor, perm);
  k_cvt_x  <<<(N_TOK * DIM) / 256, 256, 0, stream>>>(x, xbf);
  k_cvt_E  <<<NCODES, DIM, 0, stream>>>(E, Ebf, Enbf, EbfT);
  k_quant  <<<NTYPES * MAXTILES, 32, 0, stream>>>(xbf, Ebf, EbfT, counts, offsets,
                                                  perm, tau, out);
  k_uniform<<<dim3(ROWTILES, COLCHUNKS), 32, 0, stream>>>(Enbf, pSum, pPos);
  k_final  <<<1, 256, 0, stream>>>(pSum, pPos, out);
}
